// MambaBlocks_62809601736988
// MI455X (gfx1250) — compile-verified, matrix operand order checked by reading, not run on hardware
//
#include <hip/hip_runtime.h>
#include <hip/hip_bf16.h>

typedef __attribute__((ext_vector_type(16))) _Float16 v16h;
typedef __attribute__((ext_vector_type(8)))  float    v8f;

#define BATCH   2
#define SEQ     512
#define DIM     64
#define DINNER  1024
#define NSTATE  256
#define DTRANK  4
#define XDBL_W  (DTRANK + 2 * NSTATE)   // 516
#define NBLOCK  2
#define ROWS    (BATCH * SEQ)           // 1024

// ---------------------------------------------------------------------------
// WMMA tile loaders: f32 global -> f16 per-lane layout (CDNA5 ISA 7.12.2).
// A 16x32 f16: lane&15 = row M; lanes 0-15 hold K {0..7,16..23},
// lanes 16-31 hold K {8..15,24..31}; VGPR j packs (k, k+1).
// ---------------------------------------------------------------------------
__device__ __forceinline__ v16h load_a16x32(const float* __restrict__ A, int lda,
                                            int m0, int k0, int lane) {
  int r  = lane & 15;
  int kb = (lane >> 4) * 8;
  const float* row = A + (size_t)(m0 + r) * lda + k0 + kb;
  v16h v = {};
#pragma unroll
  for (int j = 0; j < 8; ++j) {
    int k = ((j & 4) ? 16 : 0) + 2 * (j & 3);
    v[2 * j]     = (_Float16)row[k];
    v[2 * j + 1] = (_Float16)row[k + 1];
  }
  return v;
}

// B 32x16: weight W stored row-major (N, K); B[k][n] = W[n][k].
// Load as transpose-of-A layout: lane&15 = column n, same K grouping.
__device__ __forceinline__ v16h load_b32x16(const float* __restrict__ W, int ldw,
                                            int n0, int k0, int nvalid, int lane) {
  int r  = lane & 15;
  int kb = (lane >> 4) * 8;
  int n  = n0 + r;
  v16h v = {};
  if (n < nvalid) {
    const float* row = W + (size_t)n * ldw + k0 + kb;
#pragma unroll
    for (int j = 0; j < 8; ++j) {
      int k = ((j & 4) ? 16 : 0) + 2 * (j & 3);
      v[2 * j]     = (_Float16)row[k];
      v[2 * j + 1] = (_Float16)row[k + 1];
    }
  }
  return v;
}

__device__ __forceinline__ void store_tile(float* __restrict__ C,
                                           const float* __restrict__ res,
                                           const v8f& acc, int m0, int n0,
                                           int N, int ldc, int lane) {
  int n = n0 + (lane & 15);
  if (n < N) {
    int mbase = m0 + ((lane >> 4) ? 8 : 0);
#pragma unroll
    for (int v = 0; v < 8; ++v) {
      size_t off = (size_t)(mbase + v) * ldc + n;
      float val = acc[v];
      if (res) val += res[off];
      C[off] = val;
    }
  }
}

// C[M,N] = A[M,K] * W[N,K]^T (+ res).
// One 32x32 macro-tile (2x2 WMMA tiles) per wave, 8 waves per block:
// 4 independent v_wmma per K-step, A/B fetches amortized 2x each.
__global__ void gemm_wmma_kernel(const float* __restrict__ A,
                                 const float* __restrict__ W,
                                 float* __restrict__ C,
                                 const float* __restrict__ res,
                                 int N, int K, int lda, int ldw, int ldc,
                                 int nsupers) {
  int lane = threadIdx.x & 31;
  int wave = threadIdx.x >> 5;
  int ns   = blockIdx.y * 8 + wave;
  if (ns >= nsupers) return;           // uniform per wave: EXEC stays all-ones
  int m0 = blockIdx.x * 32;
  int n0 = ns * 32;

  v8f acc00 = {}, acc01 = {}, acc10 = {}, acc11 = {};
  for (int k0 = 0; k0 < K; k0 += 32) {
    v16h a0 = load_a16x32(A, lda, m0,      k0, lane);
    v16h a1 = load_a16x32(A, lda, m0 + 16, k0, lane);
    v16h b0 = load_b32x16(W, ldw, n0,      k0, N, lane);
    v16h b1 = load_b32x16(W, ldw, n0 + 16, k0, N, lane);
    acc00 = __builtin_amdgcn_wmma_f32_16x16x32_f16(false, a0, false, b0,
                                                   (short)0, acc00, false, false);
    acc01 = __builtin_amdgcn_wmma_f32_16x16x32_f16(false, a0, false, b1,
                                                   (short)0, acc01, false, false);
    acc10 = __builtin_amdgcn_wmma_f32_16x16x32_f16(false, a1, false, b0,
                                                   (short)0, acc10, false, false);
    acc11 = __builtin_amdgcn_wmma_f32_16x16x32_f16(false, a1, false, b1,
                                                   (short)0, acc11, false, false);
  }

  store_tile(C, res, acc00, m0,      n0,      N, ldc, lane);
  store_tile(C, res, acc01, m0,      n0 + 16, N, ldc, lane);
  store_tile(C, res, acc10, m0 + 16, n0,      N, ldc, lane);
  store_tile(C, res, acc11, m0 + 16, n0 + 16, N, ldc, lane);
}

__device__ __forceinline__ float silu_f(float x) {
  return x * (1.f / (1.f + __expf(-x)));
}

// u = silu(causal_dwconv(xs) + cb); xs = xz[:, :, 0:DINNER]
__global__ void conv_silu_kernel(const float* __restrict__ xz,
                                 const float* __restrict__ cw,
                                 const float* __restrict__ cb,
                                 float* __restrict__ u) {
  int idx = blockIdx.x * blockDim.x + threadIdx.x;
  if (idx >= ROWS * DINNER) return;
  int d   = idx & (DINNER - 1);
  int row = idx >> 10;                 // b*SEQ + l
  int l   = row & (SEQ - 1);
  int b   = row >> 9;
  float acc = cb[d];
#pragma unroll
  for (int k = 0; k < 4; ++k) {
    int ls = l - 3 + k;
    if (ls >= 0)
      acc = fmaf(xz[((size_t)(b * SEQ + ls)) * (2 * DINNER) + d], cw[d * 4 + k], acc);
  }
  u[idx] = silu_f(acc);
}

// dt = softplus(x_dbl[:, :, 0:4] @ dtw^T + dtb)
__global__ void dt_softplus_kernel(const float* __restrict__ xdbl,
                                   const float* __restrict__ dtw,
                                   const float* __restrict__ dtb,
                                   float* __restrict__ dt) {
  int idx = blockIdx.x * blockDim.x + threadIdx.x;
  if (idx >= ROWS * DINNER) return;
  int d   = idx & (DINNER - 1);
  int row = idx >> 10;
  const float* xr = xdbl + (size_t)row * XDBL_W;
  float t = dtb[d];
#pragma unroll
  for (int r = 0; r < DTRANK; ++r) t = fmaf(xr[r], dtw[d * DTRANK + r], t);
  dt[idx] = (t > 20.f) ? t : log1pf(__expf(t));
}

// Selective scan: one wave per (b, d); lane holds 8 states n = lane + 32*j.
// Next step's B/C row is prefetched (global_prefetch_b8) to hide the latency
// of the only loads on the 512-step serial dependence chain.
__global__ void scan_kernel(const float* __restrict__ dt,
                            const float* __restrict__ u,
                            const float* __restrict__ xdbl,
                            const float* __restrict__ A_log,
                            float* __restrict__ y) {
  int lane = threadIdx.x & 31;
  int wave = threadIdx.x >> 5;
  int b = blockIdx.x / (DINNER / 8);
  int d = (blockIdx.x % (DINNER / 8)) * 8 + wave;

  float a[8], h[8];
#pragma unroll
  for (int j = 0; j < 8; ++j) {
    a[j] = -__expf(A_log[(size_t)d * NSTATE + lane + 32 * j]);
    h[j] = 0.f;
  }

  for (int l = 0; l < SEQ; ++l) {
    size_t ro  = (size_t)(b * SEQ + l);
    float dtv  = dt[ro * DINNER + d];
    float uv   = u[ro * DINNER + d];
    float dtu  = dtv * uv;
    const float* Bp = xdbl + ro * XDBL_W + DTRANK;
    const float* Cp = Bp + NSTATE;
    if (l + 1 < SEQ) {
      const float* nxt = xdbl + (ro + 1) * XDBL_W + DTRANK;
      __builtin_prefetch(nxt + lane * 4, 0, 3);
      __builtin_prefetch(nxt + NSTATE + lane * 4, 0, 3);
    }
    float acc = 0.f;
#pragma unroll
    for (int j = 0; j < 8; ++j) {
      float Bv = Bp[lane + 32 * j];
      float Cv = Cp[lane + 32 * j];
      h[j] = fmaf(h[j], __expf(dtv * a[j]), dtu * Bv);
      acc  = fmaf(h[j], Cv, acc);
    }
#pragma unroll
    for (int off = 16; off >= 1; off >>= 1) acc += __shfl_xor(acc, off, 32);
    if (lane == 0) y[ro * DINNER + d] = acc;
  }
}

// y = (y + u*D) * silu(z); z = xz[:, :, DINNER:2*DINNER]  (in place on y)
__global__ void finalize_kernel(float* __restrict__ y,
                                const float* __restrict__ u,
                                const float* __restrict__ xz,
                                const float* __restrict__ Dp) {
  int idx = blockIdx.x * blockDim.x + threadIdx.x;
  if (idx >= ROWS * DINNER) return;
  int d   = idx & (DINNER - 1);
  int row = idx >> 10;
  float z = xz[(size_t)row * (2 * DINNER) + DINNER + d];
  y[idx] = fmaf(u[idx], Dp[d], y[idx]) * silu_f(z);
}

// LayerNorm over last dim (64): one wave per row, 2 elements per lane.
__global__ void layernorm_kernel(const float* __restrict__ in,
                                 float* __restrict__ out,
                                 const float* __restrict__ g,
                                 const float* __restrict__ bb) {
  int lane = threadIdx.x & 31;
  int wave = threadIdx.x >> 5;
  int row  = blockIdx.x * 8 + wave;
  if (row >= ROWS) return;
  const float* p = in + (size_t)row * DIM;
  float x0 = p[lane], x1 = p[lane + 32];
  float s = x0 + x1;
#pragma unroll
  for (int o = 16; o >= 1; o >>= 1) s += __shfl_xor(s, o, 32);
  float mu = s * (1.f / 64.f);
  float d0 = x0 - mu, d1 = x1 - mu;
  float v = d0 * d0 + d1 * d1;
#pragma unroll
  for (int o = 16; o >= 1; o >>= 1) v += __shfl_xor(v, o, 32);
  float rstd = rsqrtf(v * (1.f / 64.f) + 1e-6f);
  out[(size_t)row * DIM + lane]      = d0 * rstd * g[lane] + bb[lane];
  out[(size_t)row * DIM + lane + 32] = d1 * rstd * g[lane + 32] + bb[lane + 32];
}

extern "C" void kernel_launch(void* const* d_in, const int* in_sizes, int n_in,
                              void* d_out, int out_size, void* d_ws, size_t ws_size,
                              hipStream_t stream) {
  const float* x_in  = (const float*)d_in[0];
  const float* inw   = (const float*)d_in[1];   // (2, 2048, 64)
  const float* cw    = (const float*)d_in[2];   // (2, 1024, 4)
  const float* cb    = (const float*)d_in[3];   // (2, 1024)
  const float* xpw   = (const float*)d_in[4];   // (2, 516, 1024)
  const float* dtw   = (const float*)d_in[5];   // (2, 1024, 4)
  const float* dtb   = (const float*)d_in[6];   // (2, 1024)
  const float* Alog  = (const float*)d_in[7];   // (2, 1024, 256)
  const float* Dp    = (const float*)d_in[8];   // (2, 1024)
  const float* outw  = (const float*)d_in[9];   // (2, 64, 1024)
  const float* lng   = (const float*)d_in[10];  // (2, 64)
  const float* lnb   = (const float*)d_in[11];  // (2, 64)

  float* ws   = (float*)d_ws;
  float* xz   = ws;                                  // ROWS * 2048
  float* u    = xz   + (size_t)ROWS * 2 * DINNER;    // ROWS * 1024
  float* xdbl = u    + (size_t)ROWS * DINNER;        // ROWS * 516
  float* dt   = xdbl + (size_t)ROWS * XDBL_W;        // ROWS * 1024
  float* ysc  = dt   + (size_t)ROWS * DINNER;        // ROWS * 1024
  float* xA   = ysc  + (size_t)ROWS * DINNER;        // ROWS * 64
  float* xB   = xA   + (size_t)ROWS * DIM;           // ROWS * 64

  const int elems = ROWS * DINNER;                   // 1,048,576
  const int eb    = (elems + 255) / 256;

  for (int i = 0; i < NBLOCK; ++i) {
    const float* xsrc = (i == 0) ? x_in : xB;
    float* xdst = (i == NBLOCK - 1) ? (float*)d_out : xA;

    const float* inw_i  = inw  + (size_t)i * 2 * DINNER * DIM;
    const float* cw_i   = cw   + (size_t)i * DINNER * 4;
    const float* cb_i   = cb   + (size_t)i * DINNER;
    const float* xpw_i  = xpw  + (size_t)i * XDBL_W * DINNER;
    const float* dtw_i  = dtw  + (size_t)i * DINNER * DTRANK;
    const float* dtb_i  = dtb  + (size_t)i * DINNER;
    const float* Alog_i = Alog + (size_t)i * DINNER * NSTATE;
    const float* Dp_i   = Dp   + (size_t)i * DINNER;
    const float* outw_i = outw + (size_t)i * DIM * DINNER;

    // xz = xsrc @ in_proj_w^T : (1024,64) x (2048,64)^T -> (1024,2048)
    // 32 m-supers x 64 n-supers
    gemm_wmma_kernel<<<dim3(ROWS / 32, 8), 256, 0, stream>>>(
        xsrc, inw_i, xz, nullptr, 2 * DINNER, DIM, DIM, DIM, 2 * DINNER, 64);

    conv_silu_kernel<<<eb, 256, 0, stream>>>(xz, cw_i, cb_i, u);

    // x_dbl = u @ x_proj_w^T : (1024,1024) x (516,1024)^T -> (1024,516)
    // 17 n-supers (last one partially/fully guarded against N=516)
    gemm_wmma_kernel<<<dim3(ROWS / 32, 3), 256, 0, stream>>>(
        u, xpw_i, xdbl, nullptr, XDBL_W, DINNER, DINNER, DINNER, XDBL_W, 17);

    dt_softplus_kernel<<<eb, 256, 0, stream>>>(xdbl, dtw_i, dtb_i, dt);

    scan_kernel<<<BATCH * (DINNER / 8), 256, 0, stream>>>(dt, u, xdbl, Alog_i, ysc);

    finalize_kernel<<<eb, 256, 0, stream>>>(ysc, u, xz, Dp_i);

    // xdst = ysc @ out_proj_w^T + xsrc : (1024,1024) x (64,1024)^T -> (1024,64)
    // 2 n-supers
    gemm_wmma_kernel<<<dim3(ROWS / 32, 1), 256, 0, stream>>>(
        ysc, outw_i, xdst, xsrc, DIM, DINNER, DINNER, DINNER, DIM, 2);

    if (i != NBLOCK - 1)
      layernorm_kernel<<<(ROWS + 7) / 8, 256, 0, stream>>>(xdst, xB,
                                                           lng + (size_t)i * DIM,
                                                           lnb + (size_t)i * DIM);
  }
  (void)in_sizes; (void)n_in; (void)out_size; (void)ws_size;
}
